// low_rank_42992622633047
// MI455X (gfx1250) — matrix operand order repaired and checked by reading, not compiled, not run on hardware
//
#include <hip/hip_runtime.h>

// Problem constants (from reference)
#define BATCH 131072
#define INK   128      // IN_SIZE (K without the bias-1 column)
#define OUTD  128      // OUT_DIM
#define RANK  4
#define MTILE 128      // batch rows per workgroup
#define PADK  136      // padded K stride (elements) in LDS: 272B rows -> conflict-free

typedef __attribute__((ext_vector_type(16))) __bf16 v16bf;
typedef __attribute__((ext_vector_type(8)))  __bf16 v8bf;
typedef __attribute__((ext_vector_type(8)))  float  v8f;

union V16 { v16bf v; v8bf h[2]; };

// LDS map (dynamic):
//   A tiles : 3 * [MTILE][PADK] bf16   (x rows, bf16, row-major)
//   B tile  : [OUTD][PADK]      bf16   (factor[r], transposed: o-major, k contiguous)
//   biasrow : [OUTD]            f32    (factor[r, 0, o])
#define A_BYTES   (3 * MTILE * PADK * 2)
#define B_BYTES   (OUTD * PADK * 2)
#define BIAS_BYTES (OUTD * 4)
#define LDS_BYTES (A_BYTES + B_BYTES + BIAS_BYTES)   // 139,776 B (< 160KB/CU)

__global__ __launch_bounds__(256)
void lmf_wmma_kernel(const float* __restrict__ x,
                     const float* __restrict__ fa,
                     const float* __restrict__ fv,
                     const float* __restrict__ ftx,
                     const float* __restrict__ fw,
                     const float* __restrict__ fb,
                     float* __restrict__ out)
{
    extern __shared__ __align__(16) char smem_raw[];
    __bf16* Alds  = (__bf16*)smem_raw;                        // [3][MTILE][PADK]
    __bf16* Blds  = (__bf16*)(smem_raw + A_BYTES);            // [OUTD][PADK]
    float*  biasl = (float*) (smem_raw + A_BYTES + B_BYTES);  // [OUTD]

    const int tid  = threadIdx.x;
    const int lane = tid & 31;
    const int wave = tid >> 5;
    const int wm   = (wave >> 1) * 32;   // wave row base within WG tile (4 groups)
    const int wn   = (wave & 1) * 64;    // wave col base (2 groups of 64 outs)
    const int g    = lane >> 4;          // lane half
    const int nloc = lane & 15;

    const int rowBlk = blockIdx.x * MTILE;

    // ---- stage x (all 3 modalities) into LDS as bf16, coalesced over k ----
    for (int m = 0; m < 3; ++m) {
        for (int idx = tid; idx < MTILE * INK; idx += 256) {
            int row = idx >> 7;          // idx / 128
            int k   = idx & 127;
            float f = x[(size_t)(rowBlk + row) * (3 * INK) + (size_t)m * INK + k];
            Alds[(m * MTILE + row) * PADK + k] = (__bf16)f;
        }
    }

    // fusion weights (uniform scalar loads)
    const float w0 = fw[0], w1 = fw[1], w2 = fw[2], w3 = fw[3];

    const v8f vzero = {};
    v8f outAcc[2][4];
    for (int i = 0; i < 2; ++i)
        for (int j = 0; j < 4; ++j) outAcc[i][j] = vzero;

    for (int r = 0; r < RANK; ++r) {
        v8f prod[2][4];
        for (int m = 0; m < 3; ++m) {
            const float* src = (m == 0) ? fa : ((m == 1) ? fv : ftx);
            src += (size_t)r * (INK + 1) * OUTD;

            __syncthreads();   // previous pass done reading Blds / A staged
            // stage factor[r][1..128][:] transposed -> Blds[o][k], bf16
            for (int idx = tid; idx < INK * OUTD; idx += 256) {
                int i = idx >> 7;        // k row
                int o = idx & 127;
                float f = src[(size_t)(i + 1) * OUTD + o];
                Blds[o * PADK + i] = (__bf16)f;
            }
            if (tid < OUTD) biasl[tid] = src[tid];   // factor[r,0,o]
            // hint: pull the next factor tile toward L2/WGP$ while we compute
            __builtin_prefetch(src + (size_t)(INK + 1) * OUTD + tid * 16, 0, 1);
            __syncthreads();

            // ---- GEMM tile: acc[ms][ns] = A_m x B  (K = 128, 4 steps of 32) ----
            v8f acc[2][4];
            for (int i = 0; i < 2; ++i)
                for (int j = 0; j < 4; ++j) acc[i][j] = vzero;

            #pragma unroll
            for (int kk = 0; kk < 4; ++kk) {
                V16 afrag[2];
                #pragma unroll
                for (int ms = 0; ms < 2; ++ms) {
                    // 16-bit A layout: lane holds row (lane%16); two 16B chunks
                    // at K = kk*32 + g*8 and kk*32 + 16 + g*8
                    int row = wm + ms * 16 + nloc;
                    const __bf16* ap = &Alds[(m * MTILE + row) * PADK + kk * 32 + g * 8];
                    afrag[ms].h[0] = *(const v8bf*)ap;
                    afrag[ms].h[1] = *(const v8bf*)(ap + 16);
                }
                #pragma unroll
                for (int ns = 0; ns < 4; ++ns) {
                    // B layout: lane holds col (lane%16); two 16B chunks
                    // at K = kk*32 + g*8 and kk*32 + 16 + g*8 (matches A)
                    V16 bfrag;
                    const __bf16* bp = &Blds[(wn + ns * 16 + nloc) * PADK + kk * 32 + g * 8];
                    bfrag.h[0] = *(const v8bf*)bp;
                    bfrag.h[1] = *(const v8bf*)(bp + 16);
                    #pragma unroll
                    for (int ms = 0; ms < 2; ++ms) {
                        acc[ms][ns] = __builtin_amdgcn_wmma_f32_16x16x32_bf16(
                            false, afrag[ms].v, false, bfrag.v,
                            (short)0, acc[ms][ns], false, false);
                    }
                }
            }

            // fold the bias-1 column: proj += factor[r,0,o]  (depends on col only)
            #pragma unroll
            for (int ns = 0; ns < 4; ++ns) {
                float bcol = biasl[wn + ns * 16 + nloc];
                #pragma unroll
                for (int ms = 0; ms < 2; ++ms)
                    #pragma unroll
                    for (int v = 0; v < 8; ++v) acc[ms][ns][v] += bcol;
            }

            // Hadamard into the running per-rank product
            if (m == 0) {
                for (int i = 0; i < 2; ++i)
                    for (int j = 0; j < 4; ++j) prod[i][j] = acc[i][j];
            } else {
                for (int i = 0; i < 2; ++i)
                    for (int j = 0; j < 4; ++j)
                        for (int v = 0; v < 8; ++v) prod[i][j][v] *= acc[i][j][v];
            }
        }
        // out += w[r] * prod
        float w = (r == 0) ? w0 : (r == 1) ? w1 : (r == 2) ? w2 : w3;
        for (int i = 0; i < 2; ++i)
            for (int j = 0; j < 4; ++j)
                for (int v = 0; v < 8; ++v) outAcc[i][j][v] += w * prod[i][j][v];
    }

    // ---- add fusion bias, store duplicated output (y2, y2) ----
    // C/D layout: lane -> col = lane%16; rows = v + 8*(lane/16) within 16-row tile
    #pragma unroll
    for (int ns = 0; ns < 4; ++ns) {
        int o   = wn + ns * 16 + nloc;
        float fbv = fb[o];
        #pragma unroll
        for (int ms = 0; ms < 2; ++ms) {
            int rbase = rowBlk + wm + ms * 16 + g * 8;
            #pragma unroll
            for (int v = 0; v < 8; ++v) {
                float val = outAcc[ms][ns][v] + fbv;
                size_t off = (size_t)(rbase + v) * OUTD + o;
                out[off] = val;
                out[(size_t)BATCH * OUTD + off] = val;
            }
        }
    }
}

extern "C" void kernel_launch(void* const* d_in, const int* in_sizes, int n_in,
                              void* d_out, int out_size, void* d_ws, size_t ws_size,
                              hipStream_t stream) {
    (void)in_sizes; (void)n_in; (void)d_ws; (void)ws_size; (void)out_size;
    const float* x  = (const float*)d_in[0];
    const float* fa = (const float*)d_in[1];
    const float* fv = (const float*)d_in[2];
    const float* ft = (const float*)d_in[3];
    const float* fw = (const float*)d_in[4];
    const float* fb = (const float*)d_in[5];
    float* out = (float*)d_out;

    // allow >64KB dynamic LDS (gfx1250 WGP has 320KB); idempotent, safe in capture
    (void)hipFuncSetAttribute((const void*)lmf_wmma_kernel,
                              hipFuncAttributeMaxDynamicSharedMemorySize, LDS_BYTES);

    dim3 grid(BATCH / MTILE);   // 1024 workgroups
    dim3 block(256);            // 8 wave32 waves: 4 row-groups x 2 col-groups
    lmf_wmma_kernel<<<grid, block, LDS_BYTES, stream>>>(x, fa, fv, ft, fw, fb, out);
}
